// ReprogrammingLayer_37589553774723
// MI455X (gfx1250) — matrix operand order repaired and checked by reading, not compiled, not run on hardware
//
#include <hip/hip_runtime.h>
#include <cstdint>

// ---------------------------------------------------------------------------
// ReprogrammingLayer on MI455X (gfx1250, wave32, WMMA f16, async LDS staging)
//   pass 0: convert activations -> f16 ; convert+transpose weights -> f16 [n][k]
//   pass 1: q/k/v projections, flash attention, out projection - all WMMA
// ---------------------------------------------------------------------------

typedef __attribute__((ext_vector_type(16))) _Float16 v16h;
typedef __attribute__((ext_vector_type(8)))  _Float16 v8h;
typedef __attribute__((ext_vector_type(8)))  float    v8f;

#define HEADS   8
#define EHEAD   128
#define DMODEL  1024
#define DLLM    4096
#define SLEN    1000
#define SPAD    1024
#define MQ      8192          // B*N
#define HE      1024          // HEADS*EHEAD

#ifndef USE_ASYNC_LDS
#define USE_ASYNC_LDS 1       // flip to 0 for synchronous staging fallback
#endif

// ---- 16B global->LDS copy: async (ASYNCcnt) or sync fallback ---------------
__device__ __forceinline__ void cp16(void* lds_dst, const void* gsrc) {
#if USE_ASYNC_LDS
  asm volatile("global_load_async_to_lds_b128 %0, %1, off"
               :
               : "v"((unsigned)(uintptr_t)lds_dst),
                 "v"((unsigned long long)(uintptr_t)gsrc)
               : "memory");
#else
  *(v8h*)lds_dst = *(const v8h*)gsrc;
#endif
}

#if USE_ASYNC_LDS
#define ASYNC_WAIT_ALL()  asm volatile("s_wait_asynccnt 0x0" ::: "memory")
#define ASYNC_WAIT_PREV() asm volatile("s_wait_asynccnt 0x6" ::: "memory")
#else
#define ASYNC_WAIT_ALL()
#define ASYNC_WAIT_PREV()
#endif

// ---- WMMA fragment helpers (ISA 7.12.2, wave32, 16-bit operands) -----------
// A 16x32: lane L<16 -> row L, K={kb+0..7, kb+16..23}, kb=0; lanes>=16: kb=8.
__device__ __forceinline__ v16h frag_a_from(const _Float16* base) {
  v8h lo = *(const v8h*)base;
  v8h hi = *(const v8h*)(base + 16);
  v16h r;
#pragma unroll
  for (int i = 0; i < 8; ++i) { r[i] = lo[i]; r[i + 8] = hi[i]; }
  return r;
}
// B 32x16: lane n<16 holds K={0..7,16..23} of col n; lanes>=16 hold K={8..15,24..31}.
// Source stored column-major [n][k].
__device__ __forceinline__ v16h frag_b_from(const _Float16* colbase) {
  v8h lo = *(const v8h*)colbase;
  v8h hi = *(const v8h*)(colbase + 16);
  v16h r;
#pragma unroll
  for (int i = 0; i < 8; ++i) { r[i] = lo[i]; r[i + 8] = hi[i]; }
  return r;
}

// ---------------------------------------------------------------------------
// Pass 0 kernels: f32 -> f16 convert, and convert + transpose for weights
// ---------------------------------------------------------------------------
__global__ __launch_bounds__(256) void cvt_f16_kernel(
    const float* __restrict__ s, _Float16* __restrict__ d, int n8) {
  const int i = blockIdx.x * 256 + threadIdx.x;
  if (i < n8) {
    const float4* s4 = (const float4*)s + (size_t)i * 2;
    const float4 a = s4[0], b = s4[1];
    v8h o;
    o[0] = (_Float16)a.x; o[1] = (_Float16)a.y;
    o[2] = (_Float16)a.z; o[3] = (_Float16)a.w;
    o[4] = (_Float16)b.x; o[5] = (_Float16)b.y;
    o[6] = (_Float16)b.z; o[7] = (_Float16)b.w;
    *((v8h*)d + i) = o;
  }
}

// W[K][Nc] f32  ->  WT[Nc][K] f16
__global__ __launch_bounds__(256) void transpose_cvt_kernel(
    const float* __restrict__ W, _Float16* __restrict__ WT, int K, int Nc) {
  __shared__ _Float16 tile[32][33];
  const int n0 = blockIdx.x * 32, k0 = blockIdx.y * 32;
  const int tx = threadIdx.x & 31, ty = threadIdx.x >> 5;
#pragma unroll
  for (int r = ty; r < 32; r += 8)
    tile[r][tx] = (_Float16)W[(size_t)(k0 + r) * Nc + n0 + tx];
  __syncthreads();
#pragma unroll
  for (int r = ty; r < 32; r += 8)
    WT[(size_t)(n0 + r) * K + k0 + tx] = tile[tx][r];
}

// ---------------------------------------------------------------------------
// f16 GEMM + bias:  C[M x Nc] = A[M x K] * WT[Nc x K]^T + bias
// BM=128 BN=64 BK=64, 256 threads = 8 waves (4 over M, 2 over N).
// Double-buffered LDS staged with async global->LDS copies.
// ---------------------------------------------------------------------------
#define BM 128
#define BN 64
#define BK 64

template <typename OT>
__global__ __launch_bounds__(256) void gemm_f16_kernel(
    const _Float16* __restrict__ A, const _Float16* __restrict__ WT,
    const float* __restrict__ bias, OT* __restrict__ C,
    int M, int K, int Nc) {
  __shared__ _Float16 lds_a[2][BM * BK];  // 2 x 16 KB
  __shared__ _Float16 lds_b[2][BN * BK];  // 2 x  8 KB

  const int t    = threadIdx.x;
  const int lane = t & 31;
  const int wave = t >> 5;
  const int wm   = wave & 3;
  const int wn   = wave >> 2;
  const int bm0  = blockIdx.y * BM;
  const int bn0  = blockIdx.x * BN;

  v8f acc[2][2];
#pragma unroll
  for (int i = 0; i < 2; ++i)
#pragma unroll
    for (int j = 0; j < 2; ++j) acc[i][j] = (v8f)0.0f;

  // stage one BMxBK A tile + BNxBK B tile into buffer p (6 x 16B per thread)
  auto stage = [&](int k0, int p) {
#pragma unroll
    for (int i = 0; i < 4; ++i) {                 // A: 1024 chunks / 256 thr
      const int c   = t + 256 * i;
      const int row = c >> 3;                     // (c*8)/64
      const int col = (c & 7) * 8;
      int gr = bm0 + row;
      gr = (gr < M) ? gr : (M - 1);               // clamp: stay in-bounds
      cp16(lds_a[p] + row * BK + col, A + (size_t)gr * K + k0 + col);
    }
#pragma unroll
    for (int i = 0; i < 2; ++i) {                 // B: 512 chunks / 256 thr
      const int c   = t + 256 * i;
      const int row = c >> 3;
      const int col = (c & 7) * 8;
      cp16(lds_b[p] + row * BK + col,
           WT + (size_t)(bn0 + row) * K + k0 + col);
    }
  };

  stage(0, 0);
  const int nst = K / BK;
  const int akb = (lane < 16) ? 0 : 8;
  const int bkb = (lane < 16) ? 0 : 8;
  const int arw = lane & 15;

  for (int ks = 0; ks < nst; ++ks) {
    const int p = ks & 1;
    if (ks + 1 < nst) {
      stage((ks + 1) * BK, p ^ 1);
      ASYNC_WAIT_PREV();                          // previous stage retired
    } else {
      ASYNC_WAIT_ALL();
    }
    // prefetch two stages ahead into L2 (global_prefetch_b8)
    if ((ks + 2) * BK < K) {
      int gr = bm0 + (t >> 1); gr = (gr < M) ? gr : (M - 1);
      __builtin_prefetch(A + (size_t)gr * K + (ks + 2) * BK, 0, 1);
    }
    __syncthreads();

#pragma unroll
    for (int sub = 0; sub < 2; ++sub) {           // BK = 2 x 32
      v16h af[2], bf[2];
#pragma unroll
      for (int i = 0; i < 2; ++i)
        af[i] = frag_a_from(lds_a[p] + (wm * 32 + i * 16 + arw) * BK +
                            sub * 32 + akb);
#pragma unroll
      for (int j = 0; j < 2; ++j)
        bf[j] = frag_b_from(lds_b[p] + (wn * 32 + j * 16 + arw) * BK +
                            sub * 32 + bkb);
#pragma unroll
      for (int i = 0; i < 2; ++i)
#pragma unroll
        for (int j = 0; j < 2; ++j)
          acc[i][j] = __builtin_amdgcn_wmma_f32_16x16x32_f16(
              false, af[i], false, bf[j], (short)0, acc[i][j], false, false);
    }
    __syncthreads();
  }

  const int off  = (lane < 16) ? 0 : 8;
  const int ncol = lane & 15;
#pragma unroll
  for (int i = 0; i < 2; ++i)
#pragma unroll
    for (int jj = 0; jj < 2; ++jj) {
      const int col = bn0 + wn * 32 + jj * 16 + ncol;
      const float b = bias[col];
#pragma unroll
      for (int j = 0; j < 8; ++j) {
        const int row = bm0 + wm * 32 + i * 16 + j + off;
        if (row < M) C[(size_t)row * Nc + col] = (OT)(acc[i][jj][j] + b);
      }
    }
}

// ---------------------------------------------------------------------------
// Flash attention: grid (MQ/64, HEADS); 4 waves/block, 16 queries per wave,
// online softmax over S in chunks of 32. K chunk staged asynchronously.
// ---------------------------------------------------------------------------
__global__ __launch_bounds__(128) void attn_kernel(
    const _Float16* __restrict__ qf, const _Float16* __restrict__ kf,
    const _Float16* __restrict__ vf, _Float16* __restrict__ of) {
  __shared__ _Float16 k_lds[32 * EHEAD];    // [s][e]   8 KB
  __shared__ _Float16 vT_lds[EHEAD * 32];   // [e][s]   8 KB
  __shared__ _Float16 p_lds[4 * 16 * 32];   // per-wave 4 KB

  const int t    = threadIdx.x;
  const int lane = t & 31;
  const int wave = t >> 5;
  const int h    = blockIdx.y;
  const int q0   = blockIdx.x * 64 + wave * 16;
  const int he0  = h * EHEAD;

  const int akb = (lane < 16) ? 0 : 8;
  const int bkb = (lane < 16) ? 0 : 8;
  const int arw = lane & 15;

  v16h aq[4];
#pragma unroll
  for (int kk = 0; kk < 4; ++kk)
    aq[kk] = frag_a_from(qf + (size_t)(q0 + arw) * HE + he0 + kk * 32 + akb);

  float mrun[8], lrun[8];
  v8f acc_o[8];
#pragma unroll
  for (int j = 0; j < 8; ++j) { mrun[j] = -1e30f; lrun[j] = 0.0f; }
#pragma unroll
  for (int e = 0; e < 8; ++e) acc_o[e] = (v8f)0.0f;

  const float scale = 0.088388347648318447f;  // 1/sqrt(128)

  for (int s0 = 0; s0 < SPAD; s0 += 32) {
    // --- K chunk [32][128]: async copies (512 x 16B / 128 threads) ---------
#pragma unroll
    for (int i = 0; i < 4; ++i) {
      const int c = t + 128 * i;
      const int s = c >> 4;
      const int e = (c & 15) * 8;
      const int gs = s0 + s;
      _Float16* dst = k_lds + s * EHEAD + e;
      if (gs < SLEN) {
        cp16(dst, kf + (size_t)gs * HE + he0 + e);
      } else {
        v8h z = {};
        *(v8h*)dst = z;
      }
    }
    // --- V chunk transposed [e][s]: synchronous (transform on the way) -----
    {
      const int s  = t >> 2;
      const int e0 = (t & 3) * 32;
      const int gs = s0 + s;
      if (gs < SLEN) {
        const _Float16* vs = vf + (size_t)gs * HE + he0 + e0;
#pragma unroll
        for (int i = 0; i < 32; ++i) vT_lds[(e0 + i) * 32 + s] = vs[i];
      } else {
#pragma unroll
        for (int i = 0; i < 32; ++i) vT_lds[(e0 + i) * 32 + s] = (_Float16)0.0f;
      }
    }
    ASYNC_WAIT_ALL();
    __syncthreads();

    // --- scores: 2 tiles x 4 K-steps = 8 WMMA ------------------------------
    v8f sc[2];
#pragma unroll
    for (int st = 0; st < 2; ++st) {
      sc[st] = (v8f)0.0f;
#pragma unroll
      for (int kk = 0; kk < 4; ++kk) {
        v16h bf = frag_b_from(k_lds + (st * 16 + arw) * EHEAD + kk * 32 + bkb);
        sc[st] = __builtin_amdgcn_wmma_f32_16x16x32_f16(
            false, aq[kk], false, bf, (short)0, sc[st], false, false);
      }
    }

#pragma unroll
    for (int st = 0; st < 2; ++st) {
      const int sidx = s0 + st * 16 + arw;
      const float msk = (sidx < SLEN) ? 0.0f : -1e30f;
#pragma unroll
      for (int j = 0; j < 8; ++j) sc[st][j] = sc[st][j] * scale + msk;
    }

    float alpha[8], mnew[8];
#pragma unroll
    for (int j = 0; j < 8; ++j) {
      float mx = fmaxf(sc[0][j], sc[1][j]);
#pragma unroll
      for (int o = 8; o >= 1; o >>= 1) mx = fmaxf(mx, __shfl_xor(mx, o, 16));
      mnew[j]  = fmaxf(mrun[j], mx);
      alpha[j] = __expf(mrun[j] - mnew[j]);
      mrun[j]  = mnew[j];
    }
#pragma unroll
    for (int st = 0; st < 2; ++st)
#pragma unroll
      for (int j = 0; j < 8; ++j) sc[st][j] = __expf(sc[st][j] - mnew[j]);
#pragma unroll
    for (int j = 0; j < 8; ++j) {
      float rs = sc[0][j] + sc[1][j];
#pragma unroll
      for (int o = 8; o >= 1; o >>= 1) rs += __shfl_xor(rs, o, 16);
      lrun[j] = lrun[j] * alpha[j] + rs;
    }
#pragma unroll
    for (int e = 0; e < 8; ++e)
#pragma unroll
      for (int j = 0; j < 8; ++j) acc_o[e][j] *= alpha[j];

    // --- repack P (C-layout -> A-layout) through per-wave LDS scratch ------
    _Float16* pw = p_lds + wave * (16 * 32);
    const int off = (lane < 16) ? 0 : 8;
#pragma unroll
    for (int st = 0; st < 2; ++st)
#pragma unroll
      for (int j = 0; j < 8; ++j)
        pw[(j + off) * 32 + st * 16 + arw] = (_Float16)sc[st][j];

    v16h pf = frag_a_from(pw + arw * 32 + akb);

#pragma unroll
    for (int e = 0; e < 8; ++e) {
      v16h bv = frag_b_from(vT_lds + (e * 16 + arw) * 32 + bkb);
      acc_o[e] = __builtin_amdgcn_wmma_f32_16x16x32_f16(
          false, pf, false, bv, (short)0, acc_o[e], false, false);
    }
    __syncthreads();
  }

  const int off = (lane < 16) ? 0 : 8;
#pragma unroll
  for (int e = 0; e < 8; ++e)
#pragma unroll
    for (int j = 0; j < 8; ++j) {
      const int row = q0 + j + off;
      const int col = he0 + e * 16 + arw;
      of[(size_t)row * HE + col] = (_Float16)(acc_o[e][j] / lrun[j]);
    }
}

// ---------------------------------------------------------------------------
extern "C" void kernel_launch(void* const* d_in, const int* in_sizes, int n_in,
                              void* d_out, int out_size, void* d_ws,
                              size_t ws_size, hipStream_t stream) {
  (void)in_sizes; (void)n_in; (void)out_size; (void)ws_size;
  const float* tgt = (const float*)d_in[0];
  const float* src = (const float*)d_in[1];
  const float* val = (const float*)d_in[2];
  const float* Wq  = (const float*)d_in[3];
  const float* bq  = (const float*)d_in[4];
  const float* Wk  = (const float*)d_in[5];
  const float* bk  = (const float*)d_in[6];
  const float* Wv  = (const float*)d_in[7];
  const float* bv  = (const float*)d_in[8];
  const float* Wo  = (const float*)d_in[9];
  const float* bo  = (const float*)d_in[10];
  float* out = (float*)d_out;

  // --- workspace layout (f16), ~98 MB total -------------------------------
  _Float16* w = (_Float16*)d_ws;
  _Float16* qf   = w; w += (size_t)MQ * HE;
  _Float16* kf   = w; w += (size_t)SPAD * HE;
  _Float16* vf   = w; w += (size_t)SPAD * HE;
  _Float16* af   = w; w += (size_t)MQ * HE;
  _Float16* tgth = w; w += (size_t)MQ * DMODEL;
  _Float16* srch = w; w += (size_t)SLEN * DLLM;
  _Float16* valh = w; w += (size_t)SLEN * DLLM;
  _Float16* wqt  = w; w += (size_t)HE * DMODEL;   // [HE][DMODEL]
  _Float16* wkt  = w; w += (size_t)HE * DLLM;     // [HE][DLLM]
  _Float16* wvt  = w; w += (size_t)HE * DLLM;     // [HE][DLLM]
  _Float16* wot  = w; w += (size_t)DLLM * HE;     // [DLLM][HE]

  const dim3 blk(256);
  // pass 0: conversions
  cvt_f16_kernel<<<(MQ * DMODEL / 8 + 255) / 256, blk, 0, stream>>>(
      tgt, tgth, MQ * DMODEL / 8);
  cvt_f16_kernel<<<(SLEN * DLLM / 8 + 255) / 256, blk, 0, stream>>>(
      src, srch, SLEN * DLLM / 8);
  cvt_f16_kernel<<<(SLEN * DLLM / 8 + 255) / 256, blk, 0, stream>>>(
      val, valh, SLEN * DLLM / 8);
  transpose_cvt_kernel<<<dim3(HE / 32, DMODEL / 32), blk, 0, stream>>>(
      Wq, wqt, DMODEL, HE);
  transpose_cvt_kernel<<<dim3(HE / 32, DLLM / 32), blk, 0, stream>>>(
      Wk, wkt, DLLM, HE);
  transpose_cvt_kernel<<<dim3(HE / 32, DLLM / 32), blk, 0, stream>>>(
      Wv, wvt, DLLM, HE);
  transpose_cvt_kernel<<<dim3(DLLM / 32, HE / 32), blk, 0, stream>>>(
      Wo, wot, HE, DLLM);

  // pass 1: projections, attention, out-projection
  gemm_f16_kernel<_Float16><<<dim3(HE / BN, MQ / BM), blk, 0, stream>>>(
      tgth, wqt, bq, qf, MQ, DMODEL, HE);
  gemm_f16_kernel<_Float16>
      <<<dim3(HE / BN, (SLEN + BM - 1) / BM), blk, 0, stream>>>(
          srch, wkt, bk, kf, SLEN, DLLM, HE);
  gemm_f16_kernel<_Float16>
      <<<dim3(HE / BN, (SLEN + BM - 1) / BM), blk, 0, stream>>>(
          valh, wvt, bv, vf, SLEN, DLLM, HE);
  attn_kernel<<<dim3(MQ / 64, HEADS), dim3(128), 0, stream>>>(qf, kf, vf, af);
  gemm_f16_kernel<float><<<dim3(DLLM / BN, MQ / BM), blk, 0, stream>>>(
      af, wot, bo, out, MQ, HE, DLLM);
}
